// ScaledDotProduct_26173530702693
// MI455X (gfx1250) — matrix operand order repaired and checked by reading, not compiled, not run on hardware
//
#include <hip/hip_runtime.h>
#include <hip/hip_bf16.h>

// Problem constants (match reference)
#define DIMD 1024
#define QK   256
#define BB   4
#define SS   4096
#define TKEY 64   // keys per attention tile

typedef __attribute__((ext_vector_type(16))) __bf16 v16bf;
typedef __attribute__((ext_vector_type(8)))  float  v8f;
typedef __attribute__((ext_vector_type(4)))  int    v4i;

// ---- CDNA5 async global->LDS path (ASYNCcnt), with portable fallback ----
#if __has_builtin(__builtin_amdgcn_global_load_async_to_lds_b128)
#define USE_ASYNC_LDS 1
// pointer-to-(v4i in AS1/AS3): the builtin takes (global int4*, shared int4*, imm, imm)
typedef __attribute__((address_space(1))) v4i* gb128p;
typedef __attribute__((address_space(3))) v4i* sb128p;
#else
#define USE_ASYNC_LDS 0
#endif

template <int N>
static __device__ inline void wait_asynccnt() {
#if USE_ASYNC_LDS
#if __has_builtin(__builtin_amdgcn_s_wait_asynccnt)
  __builtin_amdgcn_s_wait_asynccnt(N);
#else
  asm volatile("s_wait_asynccnt %0" ::"i"(N) : "memory");
#endif
#endif
}

static __device__ inline v8f zero8() {
  v8f z = {0.f, 0.f, 0.f, 0.f, 0.f, 0.f, 0.f, 0.f};
  return z;
}

static __device__ inline v8f wmma_bf16(v16bf a, v16bf b, v8f c) {
  // D = A(16x32 bf16) * B(32x16 bf16) + C(16x16 f32)
  return __builtin_amdgcn_wmma_f32_16x16x32_bf16(
      /*neg_a=*/false, a, /*neg_b=*/false, b,
      /*c_mod=*/(short)0, c, /*reuse_a=*/false, /*reuse_b=*/false);
}

// ---- A-fragment (16x32, M x K) loaders ----
// lanes 0-15 (M=lane):   K = {0..15} in elems 0..15
// lanes 16-31 (M=lane-16): K = {16..31}
static __device__ inline v16bf loadA_f32(const float* base, int ldr) {
  const int lane = threadIdx.x & 31;
  const int m  = lane & 15;
  const int hi = lane >> 4;
  const float* p = base + m * ldr + hi * 16;
  v16bf a;
#pragma unroll
  for (int i = 0; i < 8; ++i) a[i] = (__bf16)p[i];
#pragma unroll
  for (int i = 0; i < 8; ++i) a[8 + i] = (__bf16)p[8 + i];
  return a;
}

static __device__ inline v16bf loadA_bf16(const __bf16* base, int ldr) {
  const int lane = threadIdx.x & 31;
  const int m  = lane & 15;
  const int hi = lane >> 4;
  const __bf16* p = base + m * ldr + hi * 16;
  v16bf a;
#pragma unroll
  for (int i = 0; i < 8; ++i) a[i] = p[i];
#pragma unroll
  for (int i = 0; i < 8; ++i) a[8 + i] = p[8 + i];
  return a;
}

// ---- B-fragment (32x16, K x N) loaders; source row n holds B[:,n] ----
// lane%16 = N; lanes 0-15 hold K = 2j,2j+1 in VGPR j; lanes 16-31 K+16.
static __device__ inline v16bf loadBt_f32(const float* base, int ldr) {
  const int lane = threadIdx.x & 31;
  const int n  = lane & 15;
  const int hi = lane >> 4;
  const float* p = base + n * ldr + hi * 16;
  v16bf b;
#pragma unroll
  for (int i = 0; i < 16; ++i) b[i] = (__bf16)p[i];
  return b;
}

static __device__ inline v16bf loadBt_bf16(const __bf16* base, int ldr) {
  const int lane = threadIdx.x & 31;
  const int n  = lane & 15;
  const int hi = lane >> 4;
  const __bf16* p = base + n * ldr + hi * 16;
  v16bf b;
#pragma unroll
  for (int i = 0; i < 16; ++i) b[i] = p[i];
  return b;
}

// Stage one kp tile [TKEY x QK] bf16 (32 KB) into LDS. 256 threads x 128 B.
// Async path: 8 x b128 async-to-LDS per thread, tracked by ASYNCcnt.
static __device__ inline void stage_tile(const __bf16* __restrict__ kpb,
                                         __bf16* lds_dst, int kb, int tid) {
  const int srow = tid >> 2;  // 0..63 key row
  const int sseg = tid & 3;   // 4 x 128 B segments per 512 B row
  const __bf16* g = kpb + (size_t)(kb + srow) * QK + sseg * 64;
  __bf16* l = lds_dst + srow * QK + sseg * 64;
#if USE_ASYNC_LDS
#pragma unroll
  for (int i = 0; i < 8; ++i)
    __builtin_amdgcn_global_load_async_to_lds_b128(
        (gb128p)(g + i * 8), (sb128p)(l + i * 8), 0, 0);
#else
  const uint4* gs = (const uint4*)g;
  uint4* ls = (uint4*)l;
#pragma unroll
  for (int i = 0; i < 8; ++i) ls[i] = gs[i];
#endif
}

// ============================================================================
// Kernel 1: O[m, n] = sum_d X[m, d] * W[n, d] + bias[n]   (bf16 output)
// One wave computes a 16x64 output tile. If transposed!=0, output is stored
// per-batch column-major: out[(b*N + n)*SS + s] with m = b*SS + s.
// ============================================================================
__global__ __launch_bounds__(256) void proj_gemm_kernel(
    const float* __restrict__ X, const float* __restrict__ W,
    const float* __restrict__ bias, __bf16* __restrict__ out,
    int N, int transposed) {
  const int wid = blockIdx.x * 8 + (threadIdx.x >> 5);
  const int ntiles = N / 64;
  const int m0 = (wid / ntiles) * 16;
  const int n0 = (wid % ntiles) * 64;
  const int lane = threadIdx.x & 31;
  const int hi = lane >> 4;
  const int nl = lane & 15;

  v8f acc[4] = {zero8(), zero8(), zero8(), zero8()};

  for (int k0 = 0; k0 < DIMD; k0 += 32) {
    v16bf a = loadA_f32(X + (size_t)m0 * DIMD + k0, DIMD);
#pragma unroll
    for (int t = 0; t < 4; ++t) {
      v16bf b = loadBt_f32(W + (size_t)(n0 + t * 16) * DIMD + k0, DIMD);
      acc[t] = wmma_bf16(a, b, acc[t]);
    }
  }

#pragma unroll
  for (int t = 0; t < 4; ++t) {
    const int col = n0 + t * 16 + nl;
    const float bv = bias[col];
#pragma unroll
    for (int i = 0; i < 8; ++i) {
      const int row = m0 + i + hi * 8;  // C/D layout: VGPR i -> M = i (+8 hi)
      const float val = acc[t][i] + bv;
      if (!transposed) {
        out[(size_t)row * N + col] = (__bf16)val;
      } else {
        const int bidx = row >> 12;     // row / SS
        const int s    = row & (SS - 1);
        out[((size_t)bidx * N + col) * SS + s] = (__bf16)val;
      }
    }
  }
}

// ============================================================================
// Kernel 2: fused sigmoid-attention, out = sigmoid(qp kp^T / 16) @ vp.
// Block (8 waves) owns 32 query rows of one batch, all 1024 cols.
// kp tiles are double-buffered in LDS via async global->LDS loads so the
// next tile's fill overlaps the current tile's 40 WMMAs per wave.
// Wave w = (r = w>>2 row half, c = w&3 col quarter):
//   Phase A: 16x16 score subtile (K=256, 8 WMMAs from LDS kp), sigmoid,
//            stage bf16 P[32x64] in LDS.
//   Phase B: out[16 x 256] += P[16x64] * vpT (2 K-chunks x 16 N-tiles).
// ============================================================================
__global__ __launch_bounds__(256) void attn_sigmoid_kernel(
    const __bf16* __restrict__ qp, const __bf16* __restrict__ kp,
    const __bf16* __restrict__ vpT, float* __restrict__ out) {
  __shared__ __bf16 Kt[2][TKEY * QK];  // 2 x 32 KB double-buffered kp tile
  __shared__ __bf16 Pt[32 * TKEY];     // 4 KB sigmoid probabilities

  const int tid  = threadIdx.x;
  const int q0   = blockIdx.x * 32;
  const int b    = blockIdx.y;
  const int w    = tid >> 5;
  const int lane = tid & 31;
  const int r    = w >> 2;  // 0..1
  const int c    = w & 3;   // 0..3
  const int hi   = lane >> 4;
  const int nl   = lane & 15;

  const __bf16* qpb = qp  + (size_t)b * SS * QK;
  const __bf16* kpb = kp  + (size_t)b * SS * QK;
  const __bf16* vpb = vpT + (size_t)b * DIMD * SS;

  // Preload qp A-fragments for our 16 rows, full K=256 (8 chunks of 32).
  v16bf aq[8];
#pragma unroll
  for (int e = 0; e < 8; ++e)
    aq[e] = loadA_bf16(qpb + (size_t)(q0 + r * 16) * QK + e * 32, QK);

  v8f acc[16];
#pragma unroll
  for (int t = 0; t < 16; ++t) acc[t] = zero8();

  // Prologue: async-fill tile 0.
  stage_tile(kpb, &Kt[0][0], 0, tid);

  const int NIT = SS / TKEY;
  for (int it = 0; it < NIT; ++it) {
    const int kb = it * TKEY;
    const __bf16* KtCur = &Kt[it & 1][0];

    if (it + 1 < NIT) {
      // Kick off next tile, then wait only for the current one (<= 8
      // outstanding = the 8 we just issued).
      stage_tile(kpb, &Kt[(it + 1) & 1][0], kb + TKEY, tid);
      wait_asynccnt<8>();
    } else {
      wait_asynccnt<0>();
    }
    __syncthreads();  // Kt[cur] visible to all waves; prev Phase B done (Pt free)

    // ---- Phase A: scores for rows r*16.., keys (local) c*16.. ----
    v8f s = zero8();
#pragma unroll
    for (int e = 0; e < 8; ++e) {
      // B[e', key] = kp[key][e']: one LDS kp row per lane, contiguous in e'.
      v16bf bk = loadBt_bf16(KtCur + (c * 16) * QK + e * 32, QK);
      s = wmma_bf16(aq[e], bk, s);
    }
#pragma unroll
    for (int i = 0; i < 8; ++i) {
      const float z = s[i] * 0.0625f;  // 1/sqrt(QK) = 1/16
      const float p = 1.0f / (1.0f + __expf(-z));
      const int rowl = r * 16 + i + hi * 8;
      const int keyl = c * 16 + nl;
      Pt[rowl * TKEY + keyl] = (__bf16)p;
    }
    __syncthreads();  // Pt ready

    // ---- Phase B: out[16 x 256] += P[16 x 64] * vpT-slice[64 x 256] ----
#pragma unroll
    for (int kc = 0; kc < 2; ++kc) {
      v16bf ap = loadA_bf16(&Pt[(r * 16) * TKEY + kc * 32], TKEY);
#pragma unroll
      for (int t = 0; t < 16; ++t) {
        const int col0 = c * 256 + t * 16;
        // B[key, col] = vpT[col][key]: one vpT row per lane, contiguous keys.
        v16bf bv = loadBt_bf16(vpb + (size_t)col0 * SS + kb + kc * 32, SS);
        acc[t] = wmma_bf16(ap, bv, acc[t]);
      }
    }
  }

  // ---- Store f32 output [B, S, DIM] ----
#pragma unroll
  for (int t = 0; t < 16; ++t) {
    const int col = c * 256 + t * 16 + nl;
#pragma unroll
    for (int i = 0; i < 8; ++i) {
      const int row = q0 + r * 16 + i + hi * 8;
      out[((size_t)b * SS + row) * DIMD + col] = acc[t][i];
    }
  }
}

extern "C" void kernel_launch(void* const* d_in, const int* in_sizes, int n_in,
                              void* d_out, int out_size, void* d_ws, size_t ws_size,
                              hipStream_t stream) {
  const float* q  = (const float*)d_in[0];
  const float* k  = (const float*)d_in[1];
  const float* v  = (const float*)d_in[2];
  const float* Wq = (const float*)d_in[3];
  const float* bq = (const float*)d_in[4];
  const float* Wk = (const float*)d_in[5];
  const float* bk = (const float*)d_in[6];
  const float* Wv = (const float*)d_in[7];
  const float* bv = (const float*)d_in[8];
  float* out = (float*)d_out;

  // Workspace: qp[B*S*QK] bf16 (8MB) | kp[B*S*QK] bf16 (8MB) | vpT[B*DIM*S] bf16 (32MB)
  __bf16* qp  = (__bf16*)d_ws;
  __bf16* kpp = qp  + (size_t)BB * SS * QK;
  __bf16* vpT = kpp + (size_t)BB * SS * QK;

  const int M = BB * SS;  // 16-row tiles never cross a batch boundary

  {  // qp = q @ Wq^T + bq      (N = 256)
    int waves = (M / 16) * (QK / 64);
    proj_gemm_kernel<<<waves / 8, 256, 0, stream>>>(q, Wq, bq, qp, QK, 0);
  }
  {  // kp = k @ Wk^T + bk      (N = 256)
    int waves = (M / 16) * (QK / 64);
    proj_gemm_kernel<<<waves / 8, 256, 0, stream>>>(k, Wk, bk, kpp, QK, 0);
  }
  {  // vpT = (v @ Wv^T + bv)^T per batch   (N = 1024, stored transposed)
    int waves = (M / 16) * (DIMD / 64);
    proj_gemm_kernel<<<waves / 8, 256, 0, stream>>>(v, Wv, bv, vpT, DIMD, 1);
  }
  // Fused sigmoid attention
  attn_sigmoid_kernel<<<dim3(SS / 32, BB), 256, 0, stream>>>(qp, kpp, vpT, out);
}
